// RankGNN_38113539785296
// MI455X (gfx1250) — matrix operand order repaired and checked by reading, not compiled, not run on hardware
//
#include <hip/hip_runtime.h>
#include <hip/hip_bf16.h>
#include <math.h>

typedef __attribute__((ext_vector_type(16))) _Float16 v16h;
typedef __attribute__((ext_vector_type(8)))  float    v8f;

// ---------------------------------------------------------------------------
// Elementwise / utility kernels
// ---------------------------------------------------------------------------
__global__ void k_zero(float* __restrict__ p, int n) {
    int i = blockIdx.x * 256 + threadIdx.x;
    if (i < n) p[i] = 0.0f;
}

__global__ void k_deg(const int* __restrict__ ei, float* __restrict__ deg, int E) {
    int e = blockIdx.x * 256 + threadIdx.x;
    if (e < E) atomicAdd(&deg[ei[E + e]], 1.0f);
}

__global__ void k_dis(const float* __restrict__ deg, float* __restrict__ dis, int N) {
    int i = blockIdx.x * 256 + threadIdx.x;
    if (i < N) dis[i] = rsqrtf(deg[i] + 1.0f);
}

// h_next = tanh(agg + b), stride 64
__global__ void k_fin(const float* __restrict__ agg, const float* __restrict__ b,
                      float* __restrict__ h, int total) {
    int i = blockIdx.x * 256 + threadIdx.x;
    if (i < total) h[i] = tanhf(agg[i] + b[i & 63]);
}

// ---------------------------------------------------------------------------
// First layer: t = x @ W_in  (K=9, too skinny for WMMA), agg = t * dis^2
// ---------------------------------------------------------------------------
__global__ void __launch_bounds__(256)
k_in_gemm(const float* __restrict__ x, const float* __restrict__ Win,
          const float* __restrict__ dis, float* __restrict__ t,
          float* __restrict__ agg, int N) {
    __shared__ float Ws[9 * 64];
    for (int i = threadIdx.x; i < 9 * 64; i += 256) Ws[i] = Win[i];
    __syncthreads();
    int gid = blockIdx.x * 256 + threadIdx.x;
    int node = gid >> 6;
    int col  = gid & 63;
    if (node >= N) return;
    const float* xr = x + (long long)node * 9;
    float acc = 0.0f;
    #pragma unroll
    for (int k = 0; k < 9; ++k) acc += xr[k] * Ws[k * 64 + col];
    long long o = (long long)node * 64 + col;
    float ds = dis[node];
    t[o]   = acc;
    agg[o] = acc * ds * ds;
}

// ---------------------------------------------------------------------------
// WMMA GEMM: out = X(MxK=64) @ W(64 x NT*16)
//   MODE 0: outA = v ; outB = v * dis[row]^2        (GCN transform)
//   MODE 1: outA = tanh(v + bias[col])              (MLP layer)
// 8 waves/block, one 16-row tile per wave. W is pre-swizzled into LDS in the
// exact per-lane B-fragment order so each fragment is a single 32B LDS read.
// NT/MODE are compile-time -> static acc[] indexing, fully unrolled WMMAs.
// ---------------------------------------------------------------------------
template <int NT, int MODE>
__global__ void __launch_bounds__(256)
k_gemm_wmma(const float* __restrict__ X, const float* __restrict__ Wf,
            const float* __restrict__ bias, const float* __restrict__ dis,
            float* __restrict__ outA, float* __restrict__ outB, int M) {
    constexpr int NCOLS = NT * 16;
    // fragments: frag = kc*NT + nt ; per fragment 32 lanes x 16 halves (512 h)
    __shared__ _Float16 Wl[2 * NT * 512];
    const int tid = threadIdx.x;
    const int tot = 2 * NT * 512;          // == 64 * NCOLS
    for (int i = tid; i < tot; i += 256) {
        int frag = i >> 9;
        int ln   = (i >> 4) & 31;
        int j    = i & 15;
        int kc   = frag / NT;
        int nt   = frag % NT;
        int k    = kc * 32 + ((ln & 16) ? 16 : 0) + j;  // B: lanes<16 K..K+15, lanes>=16 +16
        int n    = nt * 16 + (ln & 15);
        Wl[i] = (_Float16)Wf[k * NCOLS + n];
    }
    __syncthreads();

    const int lane = tid & 31;
    const int wave = tid >> 5;
    const int m0 = (blockIdx.x * 8 + wave) * 16;
    if (m0 >= M) return;                   // wave-uniform: EXEC stays all-1s

    const int row = m0 + (lane & 15);
    const int hi  = (lane & 16) ? 1 : 0;
    const float* xr = X + (long long)row * 64;

    v8f acc[NT] = {};
    #pragma unroll
    for (int kc = 0; kc < 2; ++kc) {
        // A fragment: 16x32 f16. lanes<16: K kc*32+{0..15}; lanes>=16: +16.
        v16h a;
        const int ab = kc * 32 + hi * 16;
        #pragma unroll
        for (int j = 0; j < 8; ++j) {
            a[j]     = (_Float16)xr[ab + j];
            a[j + 8] = (_Float16)xr[ab + 8 + j];
        }
        #pragma unroll
        for (int nt = 0; nt < NT; ++nt) {
            v16h b = *(const v16h*)&Wl[(kc * NT + nt) * 512 + lane * 16];
            acc[nt] = __builtin_amdgcn_wmma_f32_16x16x32_f16(
                false, a, false, b, (short)0, acc[nt], false, false);
        }
    }

    // C/D layout: VGPR r holds M=r (lanes 0-15) / M=r+8 (lanes 16-31); N = lane&15
    const int col0  = lane & 15;
    const int rbase = hi * 8;
    #pragma unroll
    for (int nt = 0; nt < NT; ++nt) {
        #pragma unroll
        for (int r = 0; r < 8; ++r) {
            int rr = m0 + rbase + r;
            int cc = nt * 16 + col0;
            float v = acc[nt][r];
            long long o = (long long)rr * NCOLS + cc;
            if (MODE == 0) {
                float ds = dis[rr];
                outA[o] = v;
                outB[o] = v * ds * ds;
            } else {
                outA[o] = tanhf(v + bias[cc]);
            }
        }
    }
}

// ---------------------------------------------------------------------------
// Edge scatter: agg[dst] += t[src] * dis[src]*dis[dst]
// One wave per edge; float2 per lane -> coalesced 256B row; f32 atomics to L2.
// ---------------------------------------------------------------------------
__global__ void __launch_bounds__(256)
k_edge(const float* __restrict__ t, const float* __restrict__ dis,
       const int* __restrict__ ei, float* __restrict__ agg, int E) {
    int e = blockIdx.x * 8 + (threadIdx.x >> 5);
    if (e >= E) return;
    int lane = threadIdx.x & 31;
    int s = ei[e];
    int d = ei[E + e];
    float nrm = dis[s] * dis[d];
    float2 v = ((const float2*)(t + (long long)s * 64))[lane];
    float* ap = agg + (long long)d * 64 + lane * 2;
    atomicAdd(ap,     v.x * nrm);
    atomicAdd(ap + 1, v.y * nrm);
}

// ---------------------------------------------------------------------------
// fc3: u[i] = h2[i,0:32] . w + b   (stride 32 input)
// ---------------------------------------------------------------------------
__global__ void k_u(const float* __restrict__ h2, const float* __restrict__ w,
                    const float* __restrict__ b, float* __restrict__ u, int N) {
    int i = blockIdx.x * 256 + threadIdx.x;
    if (i >= N) return;
    const float* r = h2 + (long long)i * 32;
    float acc = b[0];
    #pragma unroll
    for (int k = 0; k < 32; ++k) acc += r[k] * w[k];
    u[i] = acc;
}

__global__ void k_pool(const float* __restrict__ u, const int* __restrict__ batch,
                       float* __restrict__ sums, float* __restrict__ cnts, int N) {
    int i = blockIdx.x * 256 + threadIdx.x;
    if (i >= N) return;
    int g = batch[i];
    atomicAdd(&sums[g], u[i]);
    atomicAdd(&cnts[g], 1.0f);
}

__global__ void k_xu(const float* __restrict__ sums, const float* __restrict__ cnts,
                     float* __restrict__ xu, int G) {
    int g = blockIdx.x * 256 + threadIdx.x;
    if (g < G) xu[g] = sums[g] / fmaxf(cnts[g], 1.0f);
}

__global__ void k_pairs(const float* __restrict__ xu, const int* __restrict__ ia,
                        const int* __restrict__ ib, float* __restrict__ out, int P) {
    int p = blockIdx.x * 256 + threadIdx.x;
    if (p >= P) return;
    float d = xu[ib[p]] - xu[ia[p]];
    out[p] = 1.0f / (1.0f + expf(-d));
}

// ---------------------------------------------------------------------------
// Launch
// ---------------------------------------------------------------------------
extern "C" void kernel_launch(void* const* d_in, const int* in_sizes, int n_in,
                              void* d_out, int out_size, void* d_ws, size_t ws_size,
                              hipStream_t stream) {
    const float* x     = (const float*)d_in[0];
    const int*   ei    = (const int*)d_in[1];
    const int*   batch = (const int*)d_in[2];
    const int*   idx_a = (const int*)d_in[3];
    const int*   idx_b = (const int*)d_in[4];
    const float* W_in  = (const float*)d_in[5];
    const float* b_in  = (const float*)d_in[6];
    const float* W_h   = (const float*)d_in[7];
    const float* b_h   = (const float*)d_in[8];
    const float* W_out = (const float*)d_in[9];
    const float* b_out = (const float*)d_in[10];
    const float* fc1_w = (const float*)d_in[11];
    const float* fc1_b = (const float*)d_in[12];
    const float* fc2_w = (const float*)d_in[13];
    const float* fc2_b = (const float*)d_in[14];
    const float* fc3_w = (const float*)d_in[15];
    const float* fc3_b = (const float*)d_in[16];

    const int N = in_sizes[0] / 9;        // 50000
    const int E = in_sizes[1] / 2;        // 800000
    const int P = in_sizes[3];            // 4096
    const int G = out_size - P;           // 5000

    float* out  = (float*)d_out;          // [0,P) pair outputs
    float* xu   = out + P;                // [P, P+G) x_util

    // workspace layout
    float* ws   = (float*)d_ws;
    size_t N64  = (size_t)N * 64;
    float* dis  = ws;                     // N
    float* scrN = dis + N;                // N   (deg, then u)
    float* sums = scrN + N;               // G
    float* cnts = sums + G;               // G
    float* hb   = cnts + G;               // N*64
    float* tb   = hb + N64;               // N*64
    float* ab   = tb + N64;               // N*64

    const int T = 256;
    auto blk = [](long long n, int t) { return (int)((n + t - 1) / t); };
    const int gemm_blocks = blk((N + 15) / 16, 8);   // 8 wave-tiles per block
    const int edge_blocks = blk(E, 8);               // 1 wave per edge

    // degree -> dis
    k_zero<<<blk(N, T), T, 0, stream>>>(scrN, N);
    k_zero<<<blk(G, T), T, 0, stream>>>(sums, G);
    k_zero<<<blk(G, T), T, 0, stream>>>(cnts, G);
    k_deg<<<blk(E, T), T, 0, stream>>>(ei, scrN, E);
    k_dis<<<blk(N, T), T, 0, stream>>>(scrN, dis, N);

    // layer 1: x @ W_in (K=9)
    k_in_gemm<<<blk((long long)N * 64, T), T, 0, stream>>>(x, W_in, dis, tb, ab, N);
    k_edge<<<edge_blocks, T, 0, stream>>>(tb, dis, ei, ab, E);
    k_fin<<<blk(N64, T), T, 0, stream>>>(ab, b_in, hb, (int)N64);

    // layers 2..4: WMMA transforms
    const float* Wl[3] = {W_h, W_h, W_out};
    const float* bl[3] = {b_h, b_h, b_out};
    for (int l = 0; l < 3; ++l) {
        k_gemm_wmma<4, 0><<<gemm_blocks, T, 0, stream>>>(hb, Wl[l], nullptr, dis,
                                                         tb, ab, N);
        k_edge<<<edge_blocks, T, 0, stream>>>(tb, dis, ei, ab, E);
        k_fin<<<blk(N64, T), T, 0, stream>>>(ab, bl[l], hb, (int)N64);
    }

    // fc1: tanh(h @ fc1_w + b) -> tb (N x 64)
    k_gemm_wmma<4, 1><<<gemm_blocks, T, 0, stream>>>(hb, fc1_w, fc1_b, dis,
                                                     tb, nullptr, N);
    // fc2: tanh(tb @ fc2_w + b) -> ab (N x 32)
    k_gemm_wmma<2, 1><<<gemm_blocks, T, 0, stream>>>(tb, fc2_w, fc2_b, dis,
                                                     ab, nullptr, N);
    // fc3 -> u (reuse scrN)
    k_u<<<blk(N, T), T, 0, stream>>>(ab, fc3_w, fc3_b, scrN, N);

    // mean-pool per graph, pairwise sigmoid
    k_pool<<<blk(N, T), T, 0, stream>>>(scrN, batch, sums, cnts, N);
    k_xu<<<blk(G, T), T, 0, stream>>>(sums, cnts, xu, G);
    k_pairs<<<blk(P, T), T, 0, stream>>>(xu, idx_a, idx_b, out, P);
}
